// MLP_12713103196430
// MI455X (gfx1250) — matrix operand order repaired and checked by reading, not compiled, not run on hardware
//
#include <hip/hip_runtime.h>

#define B_ROWS 32768
#define D_IN   512
#define D_HID  4096
#define D_OUT  512

typedef __bf16 bf16x16 __attribute__((ext_vector_type(16)));
typedef float  f32x8   __attribute__((ext_vector_type(8)));
typedef unsigned int u32x4 __attribute__((ext_vector_type(4)));
typedef int          i32x8 __attribute__((ext_vector_type(8)));
typedef int          i32x4 __attribute__((ext_vector_type(4)));

union Frag16 { bf16x16 v; uint4 q[2]; };

#if defined(__gfx1250__) && __has_builtin(__builtin_amdgcn_tensor_load_to_lds)
#define USE_TDM 1
#else
#define USE_TDM 0
#endif

__device__ __forceinline__ unsigned short f2bf(float f) {
  union { float f; unsigned u; } c; c.f = f;
  unsigned r = c.u + 0x7FFFu + ((c.u >> 16) & 1u);   // round-to-nearest-even
  return (unsigned short)(r >> 16);
}

// ---------- prep: xn = (x - in_mean) / in_std, packed to bf16 ----------
__global__ __launch_bounds__(256) void cvt_xn_kernel(
    const float* __restrict__ x, const float* __restrict__ in_mean,
    const float* __restrict__ in_std, unsigned short* __restrict__ xnbf) {
  const size_t base = (size_t)(blockIdx.x * 256u + threadIdx.x) * 8u;
  const int col = (int)(base & (D_IN - 1));
  const float4 a  = *(const float4*)(x + base);
  const float4 b  = *(const float4*)(x + base + 4);
  const float4 m0 = *(const float4*)(in_mean + col);
  const float4 m1 = *(const float4*)(in_mean + col + 4);
  const float4 s0 = *(const float4*)(in_std + col);
  const float4 s1 = *(const float4*)(in_std + col + 4);
  uint4 o;
  o.x = (unsigned)f2bf((a.x - m0.x) / s0.x) | ((unsigned)f2bf((a.y - m0.y) / s0.y) << 16);
  o.y = (unsigned)f2bf((a.z - m0.z) / s0.z) | ((unsigned)f2bf((a.w - m0.w) / s0.w) << 16);
  o.z = (unsigned)f2bf((b.x - m1.x) / s1.x) | ((unsigned)f2bf((b.y - m1.y) / s1.y) << 16);
  o.w = (unsigned)f2bf((b.z - m1.z) / s1.z) | ((unsigned)f2bf((b.w - m1.w) / s1.w) << 16);
  *(uint4*)(xnbf + base) = o;
}

// ---------- prep: f32 weights -> bf16 (row-major preserved) ----------
__global__ __launch_bounds__(256) void cvt_w_kernel(
    const float* __restrict__ w, unsigned short* __restrict__ wbf) {
  const size_t base = (size_t)(blockIdx.x * 256u + threadIdx.x) * 8u;
  const float4 a = *(const float4*)(w + base);
  const float4 b = *(const float4*)(w + base + 4);
  uint4 o;
  o.x = (unsigned)f2bf(a.x) | ((unsigned)f2bf(a.y) << 16);
  o.y = (unsigned)f2bf(a.z) | ((unsigned)f2bf(a.w) << 16);
  o.z = (unsigned)f2bf(b.x) | ((unsigned)f2bf(b.y) << 16);
  o.w = (unsigned)f2bf(b.z) | ((unsigned)f2bf(b.w) << 16);
  *(uint4*)(wbf + base) = o;
}

// ---------- fused MLP: 64-row tile per workgroup, 8 waves ----------
__global__ __launch_bounds__(256) void mlp_fused_kernel(
    const unsigned short* __restrict__ xnbf,
    const unsigned short* __restrict__ w1bf,   // [D_HID][D_IN] bf16
    const unsigned short* __restrict__ w2bf,   // [D_OUT][D_HID] bf16
    const float* __restrict__ b1,
    const float* __restrict__ b2,
    const float* __restrict__ out_mean,
    const float* __restrict__ out_std,
    const float* __restrict__ in_mean,
    const float* __restrict__ x,
    const int*  __restrict__ norm_max_p,
    float* __restrict__ out) {
  extern __shared__ char smem[];
  unsigned short* lds_xn = (unsigned short*)smem;            // [64][520] bf16
  unsigned short* lds_h  = (unsigned short*)(smem + 66560);  // [64][72]  bf16
  float* s_sumsq  = (float*)(smem + 75776);                  // [64]
  float* s_sumsq2 = (float*)(smem + 76032);                  // [64]

  const int tid   = threadIdx.x;
  const int lane  = tid & 31;
  const int w     = tid >> 5;
  const int laneN = lane & 15;
  const int laneH = lane >> 4;
  const int b0    = blockIdx.x * 64;

  if (tid < 64) { s_sumsq[tid] = 0.f; s_sumsq2[tid] = 0.f; }

#if USE_TDM
  // ---- Tensor Data Mover: DMA the 64x512 bf16 xn tile into LDS, adding
  //      16B of row padding (512 bf16 -> 520-element rows) via pad fields ----
  if (w == 0) {
    const unsigned long long ga =
        (unsigned long long)(const void*)(xnbf + (size_t)b0 * D_IN);
    const unsigned lds_base =
        (unsigned)(unsigned long long)(__attribute__((address_space(3))) char*)smem;
    u32x4 g0;
    g0[0] = 1u;                                   // count=1, user descriptor
    g0[1] = lds_base;                             // lds_addr (bytes)
    g0[2] = (unsigned)(ga & 0xffffffffu);         // global_addr[31:0]
    g0[3] = (unsigned)(ga >> 32) | (2u << 30);    // global_addr[56:32] | type=2
    i32x8 g1;
    g1[0] = (1 << 16)      // data_size = 2 bytes
          | (1 << 20)      // pad_enable
          | (7 << 22)      // pad_interval: 256 DWORDs (one 1KB row)
          | (3 << 25);     // pad_amount:   4 DWORDs (16 B)
    g1[1] = (int)(512u << 16);       // tensor_dim0[15:0]=512  (bits 63:48)
    g1[2] = (int)(32768u << 16);     // tensor_dim1[15:0]=32768 (bits 95:80)
    g1[3] = (int)(512u << 16);       // tile_dim0=512 (bits 127:112)
    g1[4] = 64;                      // tile_dim1=64, tile_dim2=0
    g1[5] = 512;                     // tensor_dim0_stride[31:0]=512
    g1[6] = 0;                       // stride hi / dim1_stride lo
    g1[7] = 0;
    const i32x4 gz  = {0, 0, 0, 0};                 // groups 2/3 unused (2D)
    const i32x8 gz8 = {0, 0, 0, 0, 0, 0, 0, 0};     // extra descriptor words
    __builtin_amdgcn_tensor_load_to_lds(g0, g1, gz, gz, gz8, 0);
  }
#if __has_builtin(__builtin_amdgcn_s_wait_tensorcnt)
  __builtin_amdgcn_s_wait_tensorcnt((short)0);
#else
  asm volatile("s_wait_tensorcnt 0x0" ::: "memory");
#endif
#else
  // ---- fallback: manual vector staging of the xn tile ----
  #pragma unroll
  for (int it = 0; it < 16; ++it) {
    const int id  = tid + it * 256;
    const int row = id >> 6;
    const int c8  = (id & 63) << 3;
    const uint4 v = *(const uint4*)(xnbf + (size_t)(b0 + row) * D_IN + c8);
    *(uint4*)(lds_xn + row * 520 + c8) = v;
  }
#endif
  __syncthreads();

  const int hm  = w >> 1;
  const int hn0 = (w & 1) * 2;
  const int wm  = w >> 2;
  const int wn  = w & 3;

  f32x8 oacc[2][8];
  #pragma unroll
  for (int i = 0; i < 2; ++i)
    #pragma unroll
    for (int j = 0; j < 8; ++j)
      #pragma unroll
      for (int r = 0; r < 8; ++r) oacc[i][j][r] = 0.f;

  for (int c = 0; c < 64; ++c) {
    const int hd0 = c * 64;

    // ---- GEMM1: h_chunk(64x64) = relu(xn(64x512) . W1_chunk^T + b1) ----
    f32x8 h0, h1;
    #pragma unroll
    for (int r = 0; r < 8; ++r) { h0[r] = 0.f; h1[r] = 0.f; }
    const int aRow = hm * 16 + laneN;
    const int n0   = hd0 + (hn0 + 0) * 16 + laneN;
    const int n1   = hd0 + (hn0 + 1) * 16 + laneN;
    #pragma unroll 4
    for (int kk = 0; kk < 16; ++kk) {
      const int k0 = kk * 32;
      Frag16 a;
      a.q[0] = *(const uint4*)(lds_xn + aRow * 520 + k0 + 8 * laneH);
      a.q[1] = *(const uint4*)(lds_xn + aRow * 520 + k0 + 16 + 8 * laneH);
      Frag16 bA, bB;
      const uint4* p0 = (const uint4*)(w1bf + (size_t)n0 * D_IN + k0 + 8 * laneH);
      const uint4* p1 = (const uint4*)(w1bf + (size_t)n1 * D_IN + k0 + 8 * laneH);
      bA.q[0] = p0[0]; bA.q[1] = p0[2];
      bB.q[0] = p1[0]; bB.q[1] = p1[2];
      h0 = __builtin_amdgcn_wmma_f32_16x16x32_bf16(false, a.v, false, bA.v, (short)0, h0, false, false);
      h1 = __builtin_amdgcn_wmma_f32_16x16x32_bf16(false, a.v, false, bB.v, (short)0, h1, false, false);
    }
    // bias + relu + bf16 pack into LDS h tile
    {
      const int hdl0 = (hn0 + 0) * 16 + laneN;
      const int hdl1 = (hn0 + 1) * 16 + laneN;
      const float b1v0 = b1[hd0 + hdl0];
      const float b1v1 = b1[hd0 + hdl1];
      const int mrow = hm * 16 + 8 * laneH;
      #pragma unroll
      for (int r = 0; r < 8; ++r) {
        float v0 = h0[r] + b1v0; v0 = v0 > 0.f ? v0 : 0.f;
        float v1 = h1[r] + b1v1; v1 = v1 > 0.f ? v1 : 0.f;
        lds_h[(mrow + r) * 72 + hdl0] = f2bf(v0);
        lds_h[(mrow + r) * 72 + hdl1] = f2bf(v1);
      }
    }
    // prefetch next chunk's weight rows toward L2 (global_prefetch_b8)
    if (c + 1 < 64) {
      __builtin_prefetch(w1bf + (size_t)(hd0 + 64 + (tid & 63)) * D_IN, 0, 0);
      __builtin_prefetch(w2bf + (size_t)((tid * 2) & 511) * D_HID + hd0 + 64, 0, 0);
    }
    __syncthreads();

    // ---- GEMM2: out(64x512) += h_chunk(64x64) . W2_chunk^T ----
    #pragma unroll
    for (int kk2 = 0; kk2 < 2; ++kk2) {
      const int k0 = kk2 * 32;
      const int r0 = wm * 32 + laneN;
      const int r1 = wm * 32 + 16 + laneN;
      Frag16 a0, a1;
      a0.q[0] = *(const uint4*)(lds_h + r0 * 72 + k0 + 8 * laneH);
      a0.q[1] = *(const uint4*)(lds_h + r0 * 72 + k0 + 16 + 8 * laneH);
      a1.q[0] = *(const uint4*)(lds_h + r1 * 72 + k0 + 8 * laneH);
      a1.q[1] = *(const uint4*)(lds_h + r1 * 72 + k0 + 16 + 8 * laneH);
      #pragma unroll
      for (int jn = 0; jn < 8; ++jn) {
        const int n = wn * 128 + jn * 16 + laneN;
        const uint4* pb = (const uint4*)(w2bf + (size_t)n * D_HID + hd0 + k0 + 8 * laneH);
        Frag16 bb; bb.q[0] = pb[0]; bb.q[1] = pb[2];
        oacc[0][jn] = __builtin_amdgcn_wmma_f32_16x16x32_bf16(false, a0.v, false, bb.v, (short)0, oacc[0][jn], false, false);
        oacc[1][jn] = __builtin_amdgcn_wmma_f32_16x16x32_bf16(false, a1.v, false, bb.v, (short)0, oacc[1][jn], false, false);
      }
    }
    __syncthreads();
  }

  // ---- epilogue: +b2, norm clip, affine, residual, renormalize ----
  const float nm = (float)norm_max_p[0];
  #pragma unroll
  for (int i = 0; i < 2; ++i) {
    const int mbase = wm * 32 + i * 16 + 8 * laneH;
    #pragma unroll
    for (int jn = 0; jn < 8; ++jn) {
      const int n = wn * 128 + jn * 16 + laneN;
      const float b2v = b2[n];
      #pragma unroll
      for (int r = 0; r < 8; ++r) {
        float v = oacc[i][jn][r] + b2v;
        oacc[i][jn][r] = v;
        atomicAdd(&s_sumsq[mbase + r], v * v);   // ds_add_f32
      }
    }
  }
  __syncthreads();
  #pragma unroll
  for (int i = 0; i < 2; ++i) {
    const int mbase = wm * 32 + i * 16 + 8 * laneH;
    #pragma unroll
    for (int jn = 0; jn < 8; ++jn) {
      const int n = wn * 128 + jn * 16 + laneN;
      const float ostd = out_std[n];
      const float omv  = out_mean[n];
      const float imv  = in_mean[n];
      #pragma unroll
      for (int r = 0; r < 8; ++r) {
        const int m = mbase + r;
        const float norm = sqrtf(s_sumsq[m]);
        const float sc   = (norm > nm) ? (nm / norm) : 1.f;
        float mapped = oacc[i][jn][r] * sc * ostd + omv - imv
                     + x[(size_t)(b0 + m) * D_IN + n];
        oacc[i][jn][r] = mapped;
        atomicAdd(&s_sumsq2[m], mapped * mapped);
      }
    }
  }
  __syncthreads();
  #pragma unroll
  for (int i = 0; i < 2; ++i) {
    const int mbase = wm * 32 + i * 16 + 8 * laneH;
    #pragma unroll
    for (int jn = 0; jn < 8; ++jn) {
      const int n = wn * 128 + jn * 16 + laneN;
      #pragma unroll
      for (int r = 0; r < 8; ++r) {
        const int m = mbase + r;
        float mn = sqrtf(s_sumsq2[m]);
        mn = mn > 1e-12f ? mn : 1e-12f;
        out[(size_t)(b0 + m) * D_OUT + n] = oacc[i][jn][r] / mn;
      }
    }
  }
}

extern "C" void kernel_launch(void* const* d_in, const int* in_sizes, int n_in,
                              void* d_out, int out_size, void* d_ws, size_t ws_size,
                              hipStream_t stream) {
  (void)in_sizes; (void)n_in; (void)out_size; (void)ws_size;
  const float* x        = (const float*)d_in[0];
  const float* in_mean  = (const float*)d_in[1];
  const float* in_std   = (const float*)d_in[2];
  const float* out_mean = (const float*)d_in[3];
  const float* out_std  = (const float*)d_in[4];
  const float* W1       = (const float*)d_in[5];
  const float* b1       = (const float*)d_in[6];
  const float* W2       = (const float*)d_in[7];
  const float* b2       = (const float*)d_in[8];
  const int*   nmax     = (const int*)d_in[9];
  float* out = (float*)d_out;

  unsigned short* xnbf = (unsigned short*)d_ws;                 // 32 MB
  unsigned short* w1bf = xnbf + (size_t)B_ROWS * D_IN;          // +4 MB
  unsigned short* w2bf = w1bf + (size_t)D_HID * D_IN;           // +4 MB

  cvt_xn_kernel<<<(B_ROWS * (size_t)D_IN / 8) / 256, 256, 0, stream>>>(x, in_mean, in_std, xnbf);
  cvt_w_kernel<<<((size_t)D_HID * D_IN / 8) / 256, 256, 0, stream>>>(W1, w1bf);
  cvt_w_kernel<<<((size_t)D_OUT * D_HID / 8) / 256, 256, 0, stream>>>(W2, w2bf);

  const int smem_bytes = 76288;   // xn 66560 + h 9216 + 2*256 reduction
  mlp_fused_kernel<<<B_ROWS / 64, 256, smem_bytes, stream>>>(
      xnbf, w1bf, w2bf, b1, b2, out_mean, out_std, in_mean, x, nmax, out);
}
